// DeepSeekV3Gate_38955353375115
// MI455X (gfx1250) — matrix operand order repaired and checked by reading, not compiled, not run on hardware
//
#include <hip/hip_runtime.h>
#include <hip/hip_bf16.h>

// DeepSeek-V3 sigmoid gate for MI455X (gfx1250), fused GEMM + hierarchical top-k.
// GEMM: split-precision bf16 WMMA (hi+lo) -> near-fp32 accuracy at bf16-WMMA rate.
// Fast path: weight pre-converted to bf16 hi/lo in d_ws, panels streamed with
// global_load_async_to_lds_b128 (ASYNCcnt, double-buffered) overlapping the WMMAs.

#define DIM      7168
#define NEXP     256
#define NTOK     16384
#define TOPKE    8
#define NGRP     8
#define GSZ      32
#define TOPG     4
#define RSCALE   2.5f
#define TOKTILE  64
#define NEG_INF  (-3.402823466e38f)

typedef __attribute__((ext_vector_type(4)))  float   v4f;
typedef __attribute__((ext_vector_type(8)))  float   v8f;
typedef __attribute__((ext_vector_type(8)))  __bf16  v8bf;
typedef __attribute__((ext_vector_type(16))) __bf16  v16bf;

// ---------------- common helpers ----------------
__device__ __forceinline__ v16bf frag2(const __bf16* p0, const __bf16* p1) {
    v8bf a = *(const v8bf*)p0;    // ds_load_b128
    v8bf b = *(const v8bf*)p1;    // ds_load_b128
    return __builtin_shufflevector(a, b, 0,1,2,3,4,5,6,7,8,9,10,11,12,13,14,15);
}
__device__ __forceinline__ v8f wmma_bf(v16bf a, v16bf b, v8f c) {
    return __builtin_amdgcn_wmma_f32_16x16x32_bf16(false, a, false, b, (short)0, c, false, false);
}
__device__ __forceinline__ void split8(v4f a0, v4f a1, v8bf& hv, v8bf& lv) {
#pragma unroll
    for (int i = 0; i < 4; ++i) {
        float f = a0[i]; __bf16 h = (__bf16)f; hv[i]     = h; lv[i]     = (__bf16)(f - (float)h);
        float g = a1[i]; __bf16 k = (__bf16)g; hv[i + 4] = k; lv[i + 4] = (__bf16)(g - (float)k);
    }
}
__device__ __forceinline__ float sigm(float z) { return 1.0f / (1.0f + __expf(-z)); }

// Per-token hierarchical top-k on a 64x257 LDS logits tile (threads 0..63).
__device__ __forceinline__ void topk_tail(const float (*Lt)[NEXP + 1], const float* sbias,
                                          int tok0, float* wout, int* iout) {
    const int tid = threadIdx.x;
    if (tid >= TOKTILE) return;
    const float* L = Lt[tid];
    float gs[NGRP];
#pragma unroll
    for (int g = 0; g < NGRP; ++g) {
        float m1 = NEG_INF, m2 = NEG_INF;
        for (int j = 0; j < GSZ; ++j) {
            const int e = g * GSZ + j;
            const float v = sigm(L[e]) + sbias[e];
            if (v > m1) { m2 = m1; m1 = v; }
            else if (v > m2) { m2 = v; }
        }
        gs[g] = m1 + m2;
    }
    unsigned keep = 0;
#pragma unroll
    for (int k = 0; k < TOPG; ++k) {
        int bg = 0; float bv = gs[0];
#pragma unroll
        for (int g = 1; g < NGRP; ++g) if (gs[g] > bv) { bv = gs[g]; bg = g; }
        keep |= (1u << bg);
#pragma unroll
        for (int g = 0; g < NGRP; ++g) gs[g] = (g == bg) ? NEG_INF : gs[g];
    }
    float tv[TOPKE]; int ti[TOPKE];
#pragma unroll
    for (int j = 0; j < TOPKE; ++j) { tv[j] = NEG_INF; ti[j] = 0; }
    for (int e = 0; e < NEXP; ++e) {
        if ((keep >> (e >> 5)) & 1u) {
            const float v = sigm(L[e]) + sbias[e];
            if (v > tv[TOPKE - 1]) {
                float cv = v; int ci = e;
#pragma unroll
                for (int j = 0; j < TOPKE; ++j) {
                    if (cv > tv[j]) {
                        float tf = tv[j]; int tx = ti[j];
                        tv[j] = cv; ti[j] = ci; cv = tf; ci = tx;
                    }
                }
            }
        }
    }
    float sv[TOPKE], ssum = 0.f;
#pragma unroll
    for (int j = 0; j < TOPKE; ++j) { sv[j] = sigm(L[ti[j]]); ssum += sv[j]; }
    const float sc = RSCALE / ssum;
    const int tok = tok0 + tid;
#pragma unroll
    for (int j = 0; j < TOPKE; ++j) {
        wout[(size_t)tok * TOPKE + j] = sv[j] * sc;
        iout[(size_t)tok * TOPKE + j] = ti[j];
    }
}

// ================= fast path =================
// Pre-pass: weight f32 -> bf16 hi/lo in d_ws.
__global__ __launch_bounds__(256)
void convert_w_kernel(const float* __restrict__ w, __bf16* __restrict__ whi,
                      __bf16* __restrict__ wlo) {
    const size_t i = ((size_t)blockIdx.x * blockDim.x + threadIdx.x) * 4;
    v4f f = *(const v4f*)(w + i);
#pragma unroll
    for (int j = 0; j < 4; ++j) {
        __bf16 h = (__bf16)f[j];
        whi[i + j] = h;
        wlo[i + j] = (__bf16)(f[j] - (float)h);
    }
}

#define KBF 32                 // K panel (one bf16 WMMA k-step)
#define SRF 40                 // LDS row stride in bf16 elems (20-dword stride: 16 distinct banks)
#define KITF (DIM / KBF)       // 224
#define XHI_OFF 0
#define XLO_OFF (TOKTILE * SRF * 2)                                   // 5120
#define W_OFF(buf, hl) (2 * TOKTILE * SRF * 2 + ((buf) * 2 + (hl)) * (NEXP * SRF * 2))

struct alignas(16) SmemFast {       // single __shared__ object => LDS base offset 0
    union U {
        struct S {
            __bf16 xhi[TOKTILE * SRF];     // @0
            __bf16 xlo[TOKTILE * SRF];     // @5120
            __bf16 w[2][2][NEXP * SRF];    // @10240 + (buf*2+hl)*20480, double-buffered hi/lo
        } s;                               // 92160 B
        float logits[TOKTILE][NEXP + 1];   // 65792 B
    } u;
    float sbias[NEXP];
};

__device__ __forceinline__ void async_b128(unsigned lds_off, const __bf16* gbase, int gbyte_off) {
    // GVS mode: mem = SGPR64 + VGPR32 + inst_offset ; LDS dest = wave LDS base + VGPR lds_off.
    asm volatile("global_load_async_to_lds_b128 %0, %1, %2"
                 :: "v"(lds_off), "v"(gbyte_off), "s"(gbase)
                 : "memory");
}
__device__ __forceinline__ void wait_async0() {
    asm volatile("s_wait_asynccnt 0x0" ::: "memory");
}

__global__ __launch_bounds__(256)
void dsv3_gate_fast(const float* __restrict__ x, const __bf16* __restrict__ whi_g,
                    const __bf16* __restrict__ wlo_g, const float* __restrict__ bias,
                    float* __restrict__ wout, int* __restrict__ iout) {
    __shared__ SmemFast sm;
    const int tid  = threadIdx.x;
    const int tok0 = blockIdx.x * TOKTILE;
    if (tid < NEXP) sm.sbias[tid] = bias[tid];

    const int lane = tid & 31;
    const int wave = tid >> 5;
    const int mh   = wave >> 2;        // m-tiles {2mh, 2mh+1}
    const int nq   = wave & 3;         // n-tiles {4nq .. 4nq+3}
    const int m    = lane & 15;
    const int h    = lane >> 4;

    v8f acc[2][4] = {};

    const int xr = tid >> 2, xc = tid & 3;
    const float* xsrc = x + (size_t)(tok0 + xr) * DIM + 8 * xc;

    // prologue: kick panel 0 (hi+lo) into buffer 0 via async DMA (row = tid, 4x16B chunks)
    {
        const int gb = tid * (DIM * 2);
#pragma unroll
        for (int c = 0; c < 4; ++c) {
            async_b128(W_OFF(0, 0) + tid * (SRF * 2) + c * 16, whi_g, gb + c * 16);
            async_b128(W_OFF(0, 1) + tid * (SRF * 2) + c * 16, wlo_g, gb + c * 16);
        }
    }

    for (int ki = 0; ki < KITF; ++ki) {
        const int k0  = ki * KBF;
        const int buf = ki & 1;
        // stage x panel: 8 floats/thread -> bf16 hi/lo in LDS
        {
            const float* p = xsrc + k0;
            v4f a0 = *(const v4f*)p;
            v4f a1 = *(const v4f*)(p + 4);
            v8bf hv, lv; split8(a0, a1, hv, lv);
            const int off = xr * SRF + 8 * xc;
            *(v8bf*)&sm.u.s.xhi[off] = hv;
            *(v8bf*)&sm.u.s.xlo[off] = lv;
        }
        if (ki + 1 < KITF) __builtin_prefetch(xsrc + k0 + KBF, 0, 3);
        wait_async0();           // weight panel `ki` fully landed (own waves), then...
        __syncthreads();         // ...barrier makes all waves' async data visible
        if (ki + 1 < KITF) {     // overlap: DMA next panel into the other buffer
            const int gb = tid * (DIM * 2) + (k0 + KBF) * 2;
#pragma unroll
            for (int c = 0; c < 4; ++c) {
                async_b128(W_OFF(buf ^ 1, 0) + tid * (SRF * 2) + c * 16, whi_g, gb + c * 16);
                async_b128(W_OFF(buf ^ 1, 1) + tid * (SRF * 2) + c * 16, wlo_g, gb + c * 16);
            }
        }
        // ---- WMMA: 2 m-tiles x 4 n-tiles x 3 split-precision products ----
        v16bf ahi[2], alo[2];
#pragma unroll
        for (int mi = 0; mi < 2; ++mi) {
            const int arow = ((2 * mh + mi) * 16 + m) * SRF;
            ahi[mi] = frag2(&sm.u.s.xhi[arow + 16 * h], &sm.u.s.xhi[arow + 16 * h + 8]);
            alo[mi] = frag2(&sm.u.s.xlo[arow + 16 * h], &sm.u.s.xlo[arow + 16 * h + 8]);
        }
        const __bf16* wh = sm.u.s.w[buf][0];
        const __bf16* wl = sm.u.s.w[buf][1];
#pragma unroll
        for (int t = 0; t < 4; ++t) {
            const int brow = ((4 * nq + t) * 16 + m) * SRF + 16 * h;
            v16bf bhi = frag2(&wh[brow], &wh[brow + 8]);
            v16bf blo = frag2(&wl[brow], &wl[brow + 8]);
#pragma unroll
            for (int mi = 0; mi < 2; ++mi) {
                acc[mi][t] = wmma_bf(ahi[mi], bhi, acc[mi][t]);
                acc[mi][t] = wmma_bf(alo[mi], bhi, acc[mi][t]);
                acc[mi][t] = wmma_bf(ahi[mi], blo, acc[mi][t]);
            }
        }
        __syncthreads();
    }

    // C tiles -> LDS logits (aliases stage; safe after trailing barrier)
#pragma unroll
    for (int mi = 0; mi < 2; ++mi)
#pragma unroll
        for (int t = 0; t < 4; ++t)
#pragma unroll
            for (int r = 0; r < 8; ++r)
                sm.u.logits[(2 * mh + mi) * 16 + r + 8 * h][(4 * nq + t) * 16 + m] = acc[mi][t][r];
    __syncthreads();
    topk_tail(sm.u.logits, sm.sbias, tok0, wout, iout);
}

// ================= fallback path (self-contained, no d_ws) =================
#define KBB 64
#define SRB 72
#define KITB (DIM / KBB)

struct StageB {
    __bf16 xhi[TOKTILE * SRB];
    __bf16 xlo[TOKTILE * SRB];
    __bf16 whi[NEXP * SRB];
    __bf16 wlo[NEXP * SRB];
};
union SharedB {
    StageB s;                          // 92160 B
    float logits[TOKTILE][NEXP + 1];
};

__global__ __launch_bounds__(256)
void dsv3_gate_fb(const float* __restrict__ x, const float* __restrict__ wgt,
                  const float* __restrict__ bias, float* __restrict__ wout,
                  int* __restrict__ iout) {
    __shared__ SharedB u;
    __shared__ float sbias[NEXP];
    const int tid  = threadIdx.x;
    const int tok0 = blockIdx.x * TOKTILE;
    if (tid < NEXP) sbias[tid] = bias[tid];

    const int lane = tid & 31, wave = tid >> 5;
    const int mh = wave >> 2, nq = wave & 3;
    const int m = lane & 15, h = lane >> 4;

    v8f acc[2][4] = {};

    const int xr = tid >> 2, xc = tid & 3;
    const float* xsrc = x + (size_t)(tok0 + xr) * DIM + 16 * xc;
    const float* wsrc = wgt + (size_t)tid * DIM;

    for (int ki = 0; ki < KITB; ++ki) {
        const int k0 = ki * KBB;
#pragma unroll
        for (int j = 0; j < 2; ++j) {
            const float* p = xsrc + k0 + 8 * j;
            v4f a0 = *(const v4f*)p;
            v4f a1 = *(const v4f*)(p + 4);
            v8bf hv, lv; split8(a0, a1, hv, lv);
            const int off = xr * SRB + 16 * xc + 8 * j;
            *(v8bf*)&u.s.xhi[off] = hv;
            *(v8bf*)&u.s.xlo[off] = lv;
        }
#pragma unroll
        for (int j = 0; j < 8; ++j) {
            const float* p = wsrc + k0 + 8 * j;
            v4f a0 = *(const v4f*)p;
            v4f a1 = *(const v4f*)(p + 4);
            v8bf hv, lv; split8(a0, a1, hv, lv);
            const int off = tid * SRB + 8 * j;
            *(v8bf*)&u.s.whi[off] = hv;
            *(v8bf*)&u.s.wlo[off] = lv;
        }
        if (ki + 1 < KITB) {
            __builtin_prefetch(wsrc + k0 + KBB, 0, 3);
            __builtin_prefetch(xsrc + k0 + KBB, 0, 3);
        }
        __syncthreads();
#pragma unroll
        for (int kk = 0; kk < KBB; kk += 32) {
            v16bf ahi[2], alo[2];
#pragma unroll
            for (int mi = 0; mi < 2; ++mi) {
                const int arow = ((2 * mh + mi) * 16 + m) * SRB + kk;
                ahi[mi] = frag2(&u.s.xhi[arow + 16 * h], &u.s.xhi[arow + 16 * h + 8]);
                alo[mi] = frag2(&u.s.xlo[arow + 16 * h], &u.s.xlo[arow + 16 * h + 8]);
            }
#pragma unroll
            for (int t = 0; t < 4; ++t) {
                const int brow = ((4 * nq + t) * 16 + m) * SRB + kk + 16 * h;
                v16bf bhi = frag2(&u.s.whi[brow], &u.s.whi[brow + 8]);
                v16bf blo = frag2(&u.s.wlo[brow], &u.s.wlo[brow + 8]);
#pragma unroll
                for (int mi = 0; mi < 2; ++mi) {
                    acc[mi][t] = wmma_bf(ahi[mi], bhi, acc[mi][t]);
                    acc[mi][t] = wmma_bf(alo[mi], bhi, acc[mi][t]);
                    acc[mi][t] = wmma_bf(ahi[mi], blo, acc[mi][t]);
                }
            }
        }
        __syncthreads();
    }
#pragma unroll
    for (int mi = 0; mi < 2; ++mi)
#pragma unroll
        for (int t = 0; t < 4; ++t)
#pragma unroll
            for (int r = 0; r < 8; ++r)
                u.logits[(2 * mh + mi) * 16 + r + 8 * h][(4 * nq + t) * 16 + m] = acc[mi][t][r];
    __syncthreads();
    topk_tail(u.logits, sbias, tok0, wout, iout);
}

extern "C" void kernel_launch(void* const* d_in, const int* in_sizes, int n_in,
                              void* d_out, int out_size, void* d_ws, size_t ws_size,
                              hipStream_t stream) {
    (void)in_sizes; (void)n_in; (void)out_size;
    const float* x    = (const float*)d_in[0];   // [16384, 7168]
    const float* wgt  = (const float*)d_in[1];   // [256, 7168]
    const float* bias = (const float*)d_in[2];   // [256]
    float* wout = (float*)d_out;
    int*   iout = (int*)d_out + (size_t)NTOK * TOPKE;

    const size_t need = (size_t)2 * NEXP * DIM * 2;   // whi + wlo, bf16
    if (ws_size >= need) {
        __bf16* whi = (__bf16*)d_ws;
        __bf16* wlo = whi + (size_t)NEXP * DIM;
        convert_w_kernel<<<dim3((NEXP * DIM) / (256 * 4)), dim3(256), 0, stream>>>(wgt, whi, wlo);
        dsv3_gate_fast<<<dim3(NTOK / TOKTILE), dim3(256), 0, stream>>>(x, whi, wlo, bias, wout, iout);
    } else {
        dsv3_gate_fb<<<dim3(NTOK / TOKTILE), dim3(256), 0, stream>>>(x, wgt, bias, wout, iout);
    }
}
